// GATRegressor_73108933312905
// MI455X (gfx1250) — matrix operand order repaired and checked by reading, not compiled, not run on hardware
//
#include <hip/hip_runtime.h>
#include <math.h>

#define NN 50000
#define EE 800000
#define ETOT 850000          // EE + NN self loops
#define NEG_SLOPE 0.2f

typedef _Float16 v16h __attribute__((ext_vector_type(16)));
typedef _Float16 v8h  __attribute__((ext_vector_type(8)));
typedef float    v8f  __attribute__((ext_vector_type(8)));

// ---------------------------------------------------------------- helpers
__device__ __forceinline__ void edge_nodes(const int* __restrict__ ei, int e,
                                           int& s, int& d) {
  if (e < EE) { s = ei[e]; d = ei[EE + e]; }
  else        { s = d = e - EE; }          // PyG add_self_loops
}

// monotonic uint encoding of float for atomicMax-based segment max
__device__ __forceinline__ unsigned fenc(float x) {
  unsigned b = __float_as_uint(x);
  return (b & 0x80000000u) ? ~b : (b | 0x80000000u);
}
__device__ __forceinline__ float fdec(unsigned u) {
  unsigned b = (u & 0x80000000u) ? (u ^ 0x80000000u) : ~u;
  return __uint_as_float(b);
}

// ---------------------------------------------------------------- converts
// x [NN,9] f32 -> A1 [NN,32] f16, K padded with zeros
__global__ void conv_x_pad(const float* __restrict__ x, _Float16* __restrict__ A) {
  int i = blockIdx.x * blockDim.x + threadIdx.x;
  if (i >= NN * 32) return;
  int n = i >> 5, k = i & 31;
  A[i] = (_Float16)(k < 9 ? x[n * 9 + k] : 0.0f);
}
// W1 [9,512] f32 -> W1t [512,32] f16 (transposed, K padded)
__global__ void conv_w1t(const float* __restrict__ W, _Float16* __restrict__ Wt) {
  int i = blockIdx.x * blockDim.x + threadIdx.x;
  if (i >= 512 * 32) return;
  int n = i >> 5, k = i & 31;
  Wt[i] = (_Float16)(k < 9 ? W[k * 512 + n] : 0.0f);
}
// W [K,Nout] f32 -> Wt [Nout,K] f16
__global__ void conv_wt(const float* __restrict__ W, _Float16* __restrict__ Wt,
                        int K, int Nout) {
  int i = blockIdx.x * blockDim.x + threadIdx.x;
  if (i >= K * Nout) return;
  int n = i / K, k = i % K;
  Wt[i] = (_Float16)W[k * Nout + n];
}

// ---------------------------------------------------------------- WMMA GEMM
// C[M,Nout] = A[M,K](f16,row-major) * Bt[Nout,K](f16, B column-major), f32 out.
// Each wave: 16x64 tile (4 accumulators) -> A fragment reused 4x per k-step.
// 4 waves/block tile M (64 rows), sharing the same B columns (WGP$ hits).
__global__ void wmma_gemm(const _Float16* __restrict__ A,
                          const _Float16* __restrict__ Bt,
                          float* __restrict__ C, int M, int K, int Nout) {
  const int lane = threadIdx.x & 31;
  const int wave = threadIdx.x >> 5;
  const int m0     = blockIdx.x * 64 + wave * 16;
  const int n_base = blockIdx.y * 64;
  if (m0 >= M) return;                       // wave-uniform; EXEC stays all-1s
  int nsub = (Nout - n_base) >> 4;           // #16-col subtiles this wave owns
  if (nsub > 4) nsub = 4;
  const int lm   = lane & 15;
  const int half = lane >> 4;
  const _Float16* ap = A  + (size_t)(m0 + lm) * K;
  const _Float16* bp = Bt + (size_t)(n_base + lm) * K;
  v8f acc0 = {}, acc1 = {}, acc2 = {}, acc3 = {};
  for (int k0 = 0; k0 < K; k0 += 32) {
    // A 16x32 fragment: lanes 0-15 K0..7|K16..23 ; lanes 16-31 K8..15|K24..31
    v8h a0 = *(const v8h*)(ap + k0 + half * 8);
    v8h a1 = *(const v8h*)(ap + k0 + 16 + half * 8);
    v16h av;
#pragma unroll
    for (int i = 0; i < 8; ++i) { av[i] = a0[i]; av[i + 8] = a1[i]; }
#pragma unroll
    for (int t = 0; t < 4; ++t) {
      if (t < nsub) {
        // B 32x16 fragment: lanes 0-15 K0..7|K16..23 ; lanes 16-31 K8..15|K24..31
        const _Float16* bpt = bp + (size_t)t * 16 * K;
        v8h b0 = *(const v8h*)(bpt + k0 + half * 8);
        v8h b1 = *(const v8h*)(bpt + k0 + 16 + half * 8);
        v16h bv;
#pragma unroll
        for (int i = 0; i < 8; ++i) { bv[i] = b0[i]; bv[i + 8] = b1[i]; }
        v8f& acc = (t == 0) ? acc0 : (t == 1) ? acc1 : (t == 2) ? acc2 : acc3;
        acc = __builtin_amdgcn_wmma_f32_16x16x32_f16(false, av, false, bv,
                                                     (short)0, acc, false, false);
      }
    }
  }
  // C/D layout: lane -> col lm; VGPR j -> row half*8 + j
#pragma unroll
  for (int t = 0; t < 4; ++t) {
    if (t < nsub) {
      v8f& acc = (t == 0) ? acc0 : (t == 1) ? acc1 : (t == 2) ? acc2 : acc3;
#pragma unroll
      for (int j = 0; j < 8; ++j)
        C[(size_t)(m0 + half * 8 + j) * Nout + n_base + t * 16 + lm] = acc[j];
    }
  }
}

// ---------------------------------------------------------------- attention
// es[n,h] = sum_c h[n,h*C+c]*as[h*C+c] ; ed likewise
__global__ void node_attn(const float* __restrict__ h, const float* __restrict__ as,
                          const float* __restrict__ ad, float* __restrict__ es,
                          float* __restrict__ ed, int H, int C) {
  int i = blockIdx.x * blockDim.x + threadIdx.x;
  if (i >= NN * H) return;
  int n = i / H, hd = i % H;
  const float* hp  = h  + (size_t)n * H * C + hd * C;
  const float* asp = as + hd * C;
  const float* adp = ad + hd * C;
  float s = 0.f, d = 0.f;
  for (int c = 0; c < C; ++c) { s += hp[c] * asp[c]; d += hp[c] * adp[c]; }
  es[i] = s; ed[i] = d;
}

// agg <- bias broadcast ; menc <- 0 ; den <- 0
__global__ void init_layer(float* __restrict__ agg, unsigned* __restrict__ menc,
                           float* __restrict__ den, const float* __restrict__ bias,
                           int F, int H) {
  long long i = (long long)blockIdx.x * blockDim.x + threadIdx.x;
  if (i >= (long long)NN * F) return;
  agg[i] = bias[(int)(i % F)];
  if (i < (long long)NN * H) { menc[i] = 0u; den[i] = 0.f; }
}

__global__ void edge_logit_max(const int* __restrict__ ei, const float* __restrict__ es,
                               const float* __restrict__ ed, float* __restrict__ ebuf,
                               unsigned* __restrict__ menc, int H) {
  int e = blockIdx.x * blockDim.x + threadIdx.x;
  if (e >= ETOT) return;
  int s, d; edge_nodes(ei, e, s, d);
  for (int h = 0; h < H; ++h) {
    float v = es[s * H + h] + ed[d * H + h];
    v = v > 0.f ? v : NEG_SLOPE * v;                 // leaky relu
    ebuf[(size_t)e * H + h] = v;
    atomicMax(&menc[d * H + h], fenc(v));
  }
}

__global__ void edge_exp_sum(const int* __restrict__ ei, float* __restrict__ ebuf,
                             const unsigned* __restrict__ menc, float* __restrict__ den,
                             int H) {
  int e = blockIdx.x * blockDim.x + threadIdx.x;
  if (e >= ETOT) return;
  int s, d; edge_nodes(ei, e, s, d);
  for (int h = 0; h < H; ++h) {
    float ex = __expf(ebuf[(size_t)e * H + h] - fdec(menc[d * H + h]));
    ebuf[(size_t)e * H + h] = ex;
    atomicAdd(&den[d * H + h], ex);
  }
}

__global__ void edge_alpha(const int* __restrict__ ei, float* __restrict__ ebuf,
                           const float* __restrict__ den, int H) {
  int e = blockIdx.x * blockDim.x + threadIdx.x;
  if (e >= ETOT) return;
  int s, d; edge_nodes(ei, e, s, d);
  for (int h = 0; h < H; ++h)
    ebuf[(size_t)e * H + h] /= den[d * H + h];
}

// one thread per (edge, channel): agg[dst,ch] += h[src,ch] * alpha[e, ch/C]
__global__ void scatter_agg(const int* __restrict__ ei, const float* __restrict__ h,
                            const float* __restrict__ ebuf, float* __restrict__ agg,
                            int H, int C) {
  long long i = (long long)blockIdx.x * blockDim.x + threadIdx.x;
  long long tot = (long long)ETOT * H * C;
  if (i >= tot) return;
  int F = H * C;
  int e  = (int)(i / F);
  int ch = (int)(i % F);
  int s, d; edge_nodes(ei, e, s, d);
  float a = ebuf[(size_t)e * H + ch / C];
  atomicAdd(&agg[(size_t)d * F + ch], h[(size_t)s * F + ch] * a);
}

// BN(eval, mean0/var1) + ELU, emit next layer's f16 A matrix
__global__ void post_bn_elu_f16(const float* __restrict__ agg, const float* __restrict__ g,
                                const float* __restrict__ be, _Float16* __restrict__ Aout,
                                int F) {
  long long i = (long long)blockIdx.x * blockDim.x + threadIdx.x;
  if (i >= (long long)NN * F) return;
  int c = (int)(i % F);
  float inv = rsqrtf(1.0f + 1e-5f);
  float v = agg[i] * g[c] * inv + be[c];
  v = v > 0.f ? v : (__expf(v) - 1.f);
  Aout[i] = (_Float16)v;
}

// layer-3 post + linear head: out[n] = ELU(BN(agg)) . Wh + bh
__global__ void post3_head(const float* __restrict__ agg, const float* __restrict__ g,
                           const float* __restrict__ be, const float* __restrict__ Wh,
                           const float* __restrict__ bh, float* __restrict__ out) {
  int n = blockIdx.x * blockDim.x + threadIdx.x;
  if (n >= NN) return;
  float inv = rsqrtf(1.0f + 1e-5f);
  float acc = bh[0];
#pragma unroll
  for (int c = 0; c < 32; ++c) {
    float v = agg[(size_t)n * 32 + c] * g[c] * inv + be[c];
    v = v > 0.f ? v : (__expf(v) - 1.f);
    acc += v * Wh[c];
  }
  out[n] = acc;
}

// ---------------------------------------------------------------- host side
static void gat_layer(const _Float16* A, const _Float16* Wt, int K, int H, int C,
                      const float* as, const float* ad, const float* bias,
                      float* hmat, float* es, float* ed, unsigned* menc, float* den,
                      float* ebuf, float* agg, const int* ei, hipStream_t stream) {
  const int F = H * C;
  const int t = 256;
  dim3 gg((NN + 63) / 64, (F + 63) / 64);
  wmma_gemm<<<gg, 128, 0, stream>>>(A, Wt, hmat, NN, K, F);
  node_attn<<<(NN * H + t - 1) / t, t, 0, stream>>>(hmat, as, ad, es, ed, H, C);
  long long nf = (long long)NN * F;
  init_layer<<<(unsigned)((nf + t - 1) / t), t, 0, stream>>>(agg, menc, den, bias, F, H);
  edge_logit_max<<<(ETOT + t - 1) / t, t, 0, stream>>>(ei, es, ed, ebuf, menc, H);
  edge_exp_sum<<<(ETOT + t - 1) / t, t, 0, stream>>>(ei, ebuf, menc, den, H);
  edge_alpha<<<(ETOT + t - 1) / t, t, 0, stream>>>(ei, ebuf, den, H);
  long long tot = (long long)ETOT * F;
  scatter_agg<<<(unsigned)((tot + t - 1) / t), t, 0, stream>>>(ei, hmat, ebuf, agg, H, C);
}

extern "C" void kernel_launch(void* const* d_in, const int* in_sizes, int n_in,
                              void* d_out, int out_size, void* d_ws, size_t ws_size,
                              hipStream_t stream) {
  const float* x   = (const float*)d_in[0];
  const int*   ei  = (const int*)  d_in[1];
  const float* W1  = (const float*)d_in[2];
  const float* a1s = (const float*)d_in[3];
  const float* a1d = (const float*)d_in[4];
  const float* b1  = (const float*)d_in[5];
  const float* W2  = (const float*)d_in[6];
  const float* a2s = (const float*)d_in[7];
  const float* a2d = (const float*)d_in[8];
  const float* b2  = (const float*)d_in[9];
  const float* W3  = (const float*)d_in[10];
  const float* a3s = (const float*)d_in[11];
  const float* a3d = (const float*)d_in[12];
  const float* b3  = (const float*)d_in[13];
  const float* g1  = (const float*)d_in[14];
  const float* be1 = (const float*)d_in[15];
  const float* g2  = (const float*)d_in[16];
  const float* be2 = (const float*)d_in[17];
  const float* g3  = (const float*)d_in[18];
  const float* be3 = (const float*)d_in[19];
  const float* Wh  = (const float*)d_in[20];
  const float* bh  = (const float*)d_in[21];
  float* out = (float*)d_out;

  // -------- workspace arena (256B aligned), ~242 MB total, buffers reused
  char* ws = (char*)d_ws;
  size_t o = 0;
  auto take = [&](size_t bytes) { size_t p = o; o += (bytes + 255) & ~(size_t)255; return p; };
  _Float16* A1h  = (_Float16*)(ws + take((size_t)NN * 32 * 2));
  _Float16* W1t  = (_Float16*)(ws + take((size_t)512 * 32 * 2));
  _Float16* W2t  = (_Float16*)(ws + take((size_t)128 * 512 * 2));
  _Float16* W3t  = (_Float16*)(ws + take((size_t)32 * 128 * 2));
  char*     big1 = ws + take((size_t)NN * 512 * 4);   // h1 -> later A2h|h2|agg2
  char*     big2 = ws + take((size_t)NN * 512 * 4);   // agg1 -> later A3h|h3|agg3
  float*    es   = (float*)   (ws + take((size_t)NN * 8 * 4));
  float*    ed   = (float*)   (ws + take((size_t)NN * 8 * 4));
  unsigned* menc = (unsigned*)(ws + take((size_t)NN * 8 * 4));
  float*    den  = (float*)   (ws + take((size_t)NN * 8 * 4));
  float*    ebuf = (float*)   (ws + take((size_t)ETOT * 8 * 4));

  float* h1   = (float*)big1;
  float* agg1 = (float*)big2;
  // layer-2 tensors alias big1 (h1 dead after layer-1 scatter)
  _Float16* A2h  = (_Float16*)big1;                               // NN*512 f16
  float*    h2   = (float*)(big1 + (size_t)NN * 512 * 2);         // NN*128 f32
  float*    agg2 = (float*)(big1 + (size_t)NN * 512 * 2 + (size_t)NN * 128 * 4);
  // layer-3 tensors alias big2 (agg1 dead after layer-1 post)
  _Float16* A3h  = (_Float16*)big2;                               // NN*128 f16
  float*    h3   = (float*)(big2 + (size_t)NN * 128 * 2);         // NN*32 f32
  float*    agg3 = (float*)(big2 + (size_t)NN * 128 * 2 + (size_t)NN * 32 * 4);

  const int t = 256;
  // -------- weight / input conversion to f16 (transposed B matrices)
  conv_x_pad<<<(NN * 32 + t - 1) / t, t, 0, stream>>>(x, A1h);
  conv_w1t<<<(512 * 32 + t - 1) / t, t, 0, stream>>>(W1, W1t);
  conv_wt<<<(512 * 128 + t - 1) / t, t, 0, stream>>>(W2, W2t, 512, 128);
  conv_wt<<<(128 * 32 + t - 1) / t, t, 0, stream>>>(W3, W3t, 128, 32);

  // -------- layer 1: 9->512 (8 heads x 64), concat
  gat_layer(A1h, W1t, 32, 8, 64, a1s, a1d, b1, h1, es, ed, menc, den, ebuf, agg1, ei, stream);
  long long nf1 = (long long)NN * 512;
  post_bn_elu_f16<<<(unsigned)((nf1 + t - 1) / t), t, 0, stream>>>(agg1, g1, be1, A2h, 512);

  // -------- layer 2: 512->128 (4 heads x 32), concat
  gat_layer(A2h, W2t, 512, 4, 32, a2s, a2d, b2, h2, es, ed, menc, den, ebuf, agg2, ei, stream);
  long long nf2 = (long long)NN * 128;
  post_bn_elu_f16<<<(unsigned)((nf2 + t - 1) / t), t, 0, stream>>>(agg2, g2, be2, A3h, 128);

  // -------- layer 3: 128->32 (1 head x 32), mean over 1 head == identity
  gat_layer(A3h, W3t, 128, 1, 32, a3s, a3d, b3, h3, es, ed, menc, den, ebuf, agg3, ei, stream);

  // -------- BN + ELU + linear head -> [NN,1]
  post3_head<<<(NN + t - 1) / t, t, 0, stream>>>(agg3, g3, be3, Wh, bh, out);
}
